// LinearAttention_85186381348941
// MI455X (gfx1250) — matrix operand order repaired and checked by reading, not compiled, not run on hardware
//
#include <hip/hip_runtime.h>

typedef __attribute__((ext_vector_type(16))) __bf16 v16bf;
typedef __attribute__((ext_vector_type(8)))  __bf16 v8bf;
typedef __attribute__((ext_vector_type(8)))  float  v8f;
typedef __attribute__((ext_vector_type(4)))  unsigned v4u;
typedef __attribute__((ext_vector_type(8)))  int      v8i;
typedef __attribute__((ext_vector_type(4)))  int      v4i;

#define HEADS  8
#define DIMH   64
#define NTOK   4096
#define CCH    256
#define HIDDEN 512
#define BATCH  16
#define QSCALE 0.125f   // DIM_HEAD^-0.5

#if defined(__gfx1250__) && __has_builtin(__builtin_amdgcn_tensor_load_to_lds) && \
    __has_builtin(__builtin_amdgcn_s_wait_tensorcnt)
#define HAVE_TDM 1
#else
#define HAVE_TDM 0
#endif

// ---------- helpers ----------
__device__ __forceinline__ __bf16 f2bf(float f) {
  union { float f; unsigned u; } x; x.f = f;
  unsigned r = (x.u + 0x7FFFu + ((x.u >> 16) & 1u)) >> 16;   // RNE
  union { unsigned short s; __bf16 b; } y; y.s = (unsigned short)r;
  return y.b;
}

// A fragment: 16x32 bf16, source is (M x K) row-major, 16B-aligned rows/offsets.
// ISA layout: lanes 0-15 -> row m, K {0..7,16..23}; lanes 16-31 -> K {8..15,24..31}.
__device__ __forceinline__ v16bf load_a16(const __bf16* __restrict__ p, int ld) {
  const int lane = threadIdx.x & 31;
  const int m = lane & 15, h = lane >> 4;
  const __bf16* r = p + (size_t)m * ld + h * 8;
  union { v16bf v; v8bf q[2]; } u;
  u.q[0] = *reinterpret_cast<const v8bf*>(r);
  u.q[1] = *reinterpret_cast<const v8bf*>(r + 16);
  return u.v;
}

// B fragment: 32x16 bf16, source is (N x K) row-major (i.e. B transposed).
// ISA layout: lane holds column n = lane&15; lanes 0-15 -> K {0..7,16..23}; lanes 16-31 -> K {8..15,24..31}.
__device__ __forceinline__ v16bf load_b16(const __bf16* __restrict__ p, int ld) {
  const int lane = threadIdx.x & 31;
  const int n = lane & 15, h = lane >> 4;
  const __bf16* r = p + (size_t)n * ld + h * 8;
  union { v16bf v; v8bf q[2]; } u;
  u.q[0] = *reinterpret_cast<const v8bf*>(r);
  u.q[1] = *reinterpret_cast<const v8bf*>(r + 16);
  return u.v;
}

__device__ __forceinline__ v8f wmma_bf16(v16bf a, v16bf b, v8f c) {
  return __builtin_amdgcn_wmma_f32_16x16x32_bf16(false, a, false, b, (short)0, c,
                                                 false, false);
}

#if HAVE_TDM
// TDM: async-DMA a 2D bf16 tile (tile_w x tile_h, row stride stride_elems) from
// global memory into LDS at lds_addr (row-major, packed tile_w per row).
// D# layout per CDNA5 ISA 8.3/8.4: group0 {count, lds_addr, global_addr, type=2},
// group1 {data_size=2B, tensor dims, tile dims, dim0 stride}. Groups 2/3 unused (2D).
__device__ __forceinline__ void tdm_load_tile_2d(unsigned lds_addr, const void* gptr,
                                                 unsigned tensor_w, unsigned tensor_h,
                                                 unsigned tile_w, unsigned tile_h,
                                                 unsigned stride_elems) {
  unsigned long long ga = (unsigned long long)gptr;
  v4u g0;
  g0[0] = 1u;                                                 // count = 1 valid D#
  g0[1] = lds_addr;                                           // LDS byte address
  g0[2] = (unsigned)ga;                                       // global addr lo
  g0[3] = ((unsigned)(ga >> 32) & 0x01FFFFFFu) | 0x80000000u; // addr hi | type=2
  v8i g1;
  g1[0] = (int)(1u << 16);                                    // data_size = 2 bytes
  g1[1] = (int)((tensor_w & 0xFFFFu) << 16);                  // tensor_dim0[15:0]
  g1[2] = (int)((tensor_w >> 16) | ((tensor_h & 0xFFFFu) << 16));
  g1[3] = (int)((tensor_h >> 16) | (tile_w << 16));           // tile_dim0
  g1[4] = (int)tile_h;                                        // tile_dim1 (tile_dim2=0)
  g1[5] = (int)stride_elems;                                  // tensor_dim0_stride lo
  g1[6] = 0;
  g1[7] = 0;
  v4i z4 = {0, 0, 0, 0};
#if __clang_major__ >= 23
  v8i z8 = {0, 0, 0, 0, 0, 0, 0, 0};
  __builtin_amdgcn_tensor_load_to_lds(g0, g1, z4, z4, z8, 0);
#else
  __builtin_amdgcn_tensor_load_to_lds(g0, g1, z4, z4, 0);
#endif
}
#endif

// ---------- fp32 -> bf16 elementwise ----------
__global__ void cvt_bf16_kernel(const float* __restrict__ src,
                                __bf16* __restrict__ dst, int n) {
  int i = blockIdx.x * blockDim.x + threadIdx.x;
  int stride = gridDim.x * blockDim.x;
  for (; i < n; i += stride) dst[i] = f2bf(src[i]);
}

// ---------- x (b,c,n) fp32 -> xT (b,n,c) bf16 ----------
__global__ void transpose_x_kernel(const float* __restrict__ x,
                                   __bf16* __restrict__ xT) {
  __shared__ float tile[32][33];
  const int b = blockIdx.z;
  const int n0 = blockIdx.x * 32, c0 = blockIdx.y * 32;
  const int tx = threadIdx.x, ty = threadIdx.y;
  const float* xb = x + (size_t)b * CCH * NTOK;
#pragma unroll
  for (int k2 = 0; k2 < 32; k2 += 8)
    tile[ty + k2][tx] = xb[(size_t)(c0 + ty + k2) * NTOK + n0 + tx];
  __syncthreads();
  __bf16* xtb = xT + (size_t)b * NTOK * CCH;
#pragma unroll
  for (int k2 = 0; k2 < 32; k2 += 8)
    xtb[(size_t)(n0 + ty + k2) * CCH + c0 + tx] = f2bf(tile[tx][ty + k2]);
}

// ---------- GEMM1: qkv = w_qkv(1536x256) x x(b)(256x4096) ----------
// macro-tile 128(M) x 64(N), 8 waves, each wave 2x2 16x16 tiles, K-loop 256/32.
// q third  -> qT (b,h,n,d)  fp32 (transposed store)
// k third  -> k  (b,h,d,n)  fp32
// v third  -> v  (b,h,e,n)  bf16 (direct convert)
__global__ void __launch_bounds__(256)
gemm_qkv_kernel(const __bf16* __restrict__ wq, const __bf16* __restrict__ xT,
                float* __restrict__ qT, float* __restrict__ kbuf,
                __bf16* __restrict__ vbf) {
  const int b = blockIdx.z;
  const int w = threadIdx.x >> 5;
  const int wm = w & 3, wn = w >> 2;
  const int mbase = blockIdx.y * 128 + wm * 32;
  const int nbase = blockIdx.x * 64 + wn * 32;
  const __bf16* xb = xT + (size_t)b * NTOK * CCH;
  v8f acc[2][2] = {};
  for (int kk = 0; kk < CCH; kk += 32) {
    v16bf a0 = load_a16(wq + (size_t)mbase * CCH + kk, CCH);
    v16bf a1 = load_a16(wq + (size_t)(mbase + 16) * CCH + kk, CCH);
    v16bf b0 = load_b16(xb + (size_t)nbase * CCH + kk, CCH);
    v16bf b1 = load_b16(xb + (size_t)(nbase + 16) * CCH + kk, CCH);
    acc[0][0] = wmma_bf16(a0, b0, acc[0][0]);
    acc[0][1] = wmma_bf16(a0, b1, acc[0][1]);
    acc[1][0] = wmma_bf16(a1, b0, acc[1][0]);
    acc[1][1] = wmma_bf16(a1, b1, acc[1][1]);
  }
  const int lane = threadIdx.x & 31, ln = lane & 15, lh = lane >> 4;
#pragma unroll
  for (int mt = 0; mt < 2; ++mt) {
    const int row0 = mbase + mt * 16;
#pragma unroll
    for (int nt = 0; nt < 2; ++nt) {
      const int col0 = nbase + nt * 16;
      v8f a = acc[mt][nt];
      if (row0 < HIDDEN) {                 // ---- q: transposed fp32 store
        const int hh = row0 >> 6;
        const int d0 = (row0 & 63) + lh * 8;
        float* dst = qT + (((size_t)(b * HEADS + hh) * NTOK) + col0 + ln) * DIMH + d0;
        float4 lo = {a[0], a[1], a[2], a[3]};
        float4 hi = {a[4], a[5], a[6], a[7]};
        *reinterpret_cast<float4*>(dst) = lo;
        *reinterpret_cast<float4*>(dst + 4) = hi;
      } else if (row0 < 2 * HIDDEN) {      // ---- k: (d,n) fp32 store
        const int o = row0 - HIDDEN;
        const int hh = o >> 6;
        float* dst = kbuf + ((size_t)(b * HEADS + hh) * DIMH + (o & 63)) * NTOK + col0;
#pragma unroll
        for (int r = 0; r < 8; ++r) dst[(size_t)(lh * 8 + r) * NTOK + ln] = a[r];
      } else {                             // ---- v: (e,n) bf16 store
        const int o = row0 - 2 * HIDDEN;
        const int hh = o >> 6;
        __bf16* dst = vbf + ((size_t)(b * HEADS + hh) * DIMH + (o & 63)) * NTOK + col0;
#pragma unroll
        for (int r = 0; r < 8; ++r) dst[(size_t)(lh * 8 + r) * NTOK + ln] = f2bf(a[r]);
      }
    }
  }
}

// ---------- softmax over n (4096) per (b,h,d) row, fp32 -> bf16 ----------
__global__ void __launch_bounds__(256)
softmax_k_kernel(const float* __restrict__ kin, __bf16* __restrict__ kout) {
  __shared__ float red[256];
  const size_t row = blockIdx.x;
  const float* src = kin + row * NTOK;
  __bf16* dst = kout + row * NTOK;
  const int t = threadIdx.x;
  float vals[16];
  float m = -3.4e38f;
#pragma unroll
  for (int i = 0; i < 16; ++i) { vals[i] = src[t + i * 256]; m = fmaxf(m, vals[i]); }
  red[t] = m; __syncthreads();
  for (int s = 128; s > 0; s >>= 1) {
    if (t < s) red[t] = fmaxf(red[t], red[t + s]);
    __syncthreads();
  }
  m = red[0]; __syncthreads();
  float sum = 0.f;
#pragma unroll
  for (int i = 0; i < 16; ++i) { vals[i] = __expf(vals[i] - m); sum += vals[i]; }
  red[t] = sum; __syncthreads();
  for (int s = 128; s > 0; s >>= 1) {
    if (t < s) red[t] += red[t + s];
    __syncthreads();
  }
  const float inv = 1.0f / red[0];
#pragma unroll
  for (int i = 0; i < 16; ++i) dst[t + i * 256] = f2bf(vals[i] * inv);
}

// ---------- softmax over d (64) per (b,h,n) row + SCALE, fp32 -> bf16 ----------
__global__ void __launch_bounds__(256)
softmax_q_kernel(const float* __restrict__ qin, __bf16* __restrict__ qout) {
  const int warp = threadIdx.x >> 5, lane = threadIdx.x & 31;
  const size_t row = (size_t)blockIdx.x * 8 + warp;
  const float2* src = reinterpret_cast<const float2*>(qin + row * DIMH);
  float2 v = src[lane];
  float m = fmaxf(v.x, v.y);
#pragma unroll
  for (int s = 16; s > 0; s >>= 1) m = fmaxf(m, __shfl_xor(m, s, 32));
  float e0 = __expf(v.x - m), e1 = __expf(v.y - m);
  float sum = e0 + e1;
#pragma unroll
  for (int s = 16; s > 0; s >>= 1) sum += __shfl_xor(sum, s, 32);
  const float inv = QSCALE / sum;
  __bf16* dst = qout + row * DIMH + lane * 2;
  dst[0] = f2bf(e0 * inv);
  dst[1] = f2bf(e1 * inv);
}

// ---------- GEMM2: context(d,e) = k'(64x4096) x v^T, per (b,h) ----------
// TDM path: wave0 streams double-buffered 64x32 k/v tiles into LDS with
// tensor_load_to_lds; all 4 waves compute from LDS (ds_load fragments).
// Fallback: direct global-load fragments.  Store ctx^T (e,d) bf16.
__global__ void __launch_bounds__(128)
gemm_ctx_kernel(const __bf16* __restrict__ kbf, const __bf16* __restrict__ vbf,
                __bf16* __restrict__ ctxT) {
  const int bh = blockIdx.x;
  const __bf16* kb = kbf + (size_t)bh * DIMH * NTOK;
  const __bf16* vb = vbf + (size_t)bh * DIMH * NTOK;
  const int w = threadIdx.x >> 5;
  const int mo = w * 16;
  v8f acc[4] = {};
#if HAVE_TDM
  __shared__ __bf16 kt[2][DIMH * 32];
  __shared__ __bf16 vt[2][DIMH * 32];
  const bool issuer = (w == 0);
  if (issuer) {
    tdm_load_tile_2d((unsigned)(unsigned long long)&kt[0][0], kb,
                     NTOK, DIMH, 32, DIMH, NTOK);
    tdm_load_tile_2d((unsigned)(unsigned long long)&vt[0][0], vb,
                     NTOK, DIMH, 32, DIMH, NTOK);
  }
  for (int kk = 0; kk < NTOK; kk += 32) {
    const int cur = (kk >> 5) & 1, nxt = cur ^ 1;
    if (issuer) __builtin_amdgcn_s_wait_tensorcnt(0);  // cur tiles landed
    __syncthreads();
    if (issuer && (kk + 32) < NTOK) {                  // overlap next DMA w/ math
      tdm_load_tile_2d((unsigned)(unsigned long long)&kt[nxt][0], kb + kk + 32,
                       NTOK, DIMH, 32, DIMH, NTOK);
      tdm_load_tile_2d((unsigned)(unsigned long long)&vt[nxt][0], vb + kk + 32,
                       NTOK, DIMH, 32, DIMH, NTOK);
    }
    v16bf a = load_a16(&kt[cur][mo * 32], 32);
#pragma unroll
    for (int nt = 0; nt < 4; ++nt) {
      v16bf bb = load_b16(&vt[cur][nt * 16 * 32], 32);
      acc[nt] = wmma_bf16(a, bb, acc[nt]);
    }
    __syncthreads();
  }
#else
  for (int kk = 0; kk < NTOK; kk += 32) {
    v16bf a = load_a16(kb + (size_t)mo * NTOK + kk, NTOK);
#pragma unroll
    for (int nt = 0; nt < 4; ++nt) {
      v16bf bb = load_b16(vb + (size_t)(nt * 16) * NTOK + kk, NTOK);
      acc[nt] = wmma_bf16(a, bb, acc[nt]);
    }
  }
#endif
  const int lane = threadIdx.x & 31, ln = lane & 15, lh = lane >> 4;
  __bf16* ct = ctxT + (size_t)bh * DIMH * DIMH;
#pragma unroll
  for (int nt = 0; nt < 4; ++nt) {
    v8bf pk;
#pragma unroll
    for (int r = 0; r < 8; ++r) pk[r] = f2bf(acc[nt][r]);
    *reinterpret_cast<v8bf*>(ct + (size_t)(nt * 16 + ln) * DIMH + mo + lh * 8) = pk;
  }
}

// ---------- GEMM3: out(e,n) = ctx^T(64x64) x q'(64xN), per (b,h) ----------
__global__ void __launch_bounds__(256)
gemm_out_kernel(const __bf16* __restrict__ ctxT, const __bf16* __restrict__ qbf,
                __bf16* __restrict__ outT) {
  const int bh = blockIdx.y;
  const int b = bh >> 3, hh = bh & 7;
  const __bf16* ct = ctxT + (size_t)bh * DIMH * DIMH;
  const __bf16* qb = qbf + (size_t)bh * NTOK * DIMH;
  const int n0 = blockIdx.x * 128 + (threadIdx.x >> 5) * 16;
  v8f acc[4] = {};
#pragma unroll
  for (int kk = 0; kk < DIMH; kk += 32) {
    v16bf bb = load_b16(qb + (size_t)n0 * DIMH + kk, DIMH);
#pragma unroll
    for (int mt = 0; mt < 4; ++mt) {
      v16bf a = load_a16(ct + (size_t)(mt * 16) * DIMH + kk, DIMH);
      acc[mt] = wmma_bf16(a, bb, acc[mt]);
    }
  }
  const int lane = threadIdx.x & 31, ln = lane & 15, lh = lane >> 4;
  __bf16* dst = outT + ((size_t)b * NTOK + n0 + ln) * HIDDEN + hh * DIMH;
#pragma unroll
  for (int mt = 0; mt < 4; ++mt) {
    v8bf pk;
#pragma unroll
    for (int r = 0; r < 8; ++r) pk[r] = f2bf(acc[mt][r]);
    *reinterpret_cast<v8bf*>(dst + mt * 16 + lh * 8) = pk;
  }
}

// ---------- GEMM4: y = w_out(256x512) x out(512xN) + b, fused channel L2-norm ----------
__global__ void __launch_bounds__(256)
gemm_proj_norm_kernel(const __bf16* __restrict__ wo, const __bf16* __restrict__ outT,
                      const float* __restrict__ bias, const float* __restrict__ g,
                      float* __restrict__ y) {
  __shared__ float g_l[CCH], b_l[CCH], red[8][16];
  const int t = threadIdx.x;
  g_l[t] = g[t];
  b_l[t] = bias[t];
  __syncthreads();
  const int b = blockIdx.y;
  const int w = t >> 5, lane = t & 31, ln = lane & 15, lh = lane >> 4;
  const int wm = w & 3, wn = w >> 2;
  const int n_glob = blockIdx.x * 32 + wn * 16 + ln;
  const __bf16* ob = outT + ((size_t)b * NTOK + blockIdx.x * 32 + wn * 16) * HIDDEN;
  v8f acc[4] = {};
  for (int kk = 0; kk < HIDDEN; kk += 32) {
    v16bf bb = load_b16(ob + kk, HIDDEN);
#pragma unroll
    for (int mt = 0; mt < 4; ++mt) {
      v16bf a = load_a16(wo + (size_t)(wm * 64 + mt * 16) * HIDDEN + kk, HIDDEN);
      acc[mt] = wmma_bf16(a, bb, acc[mt]);
    }
  }
  float ss = 0.f;
  float vals[4][8];
#pragma unroll
  for (int mt = 0; mt < 4; ++mt)
#pragma unroll
    for (int r = 0; r < 8; ++r) {
      const int c = wm * 64 + mt * 16 + lh * 8 + r;
      const float v = acc[mt][r] + b_l[c];
      vals[mt][r] = v;
      ss += v * v;
    }
  ss += __shfl_xor(ss, 16, 32);           // join the two 8-row halves of each column
  if (lane < 16) red[w][ln] = ss;
  __syncthreads();
  const float tot = red[wn * 4 + 0][ln] + red[wn * 4 + 1][ln] +
                    red[wn * 4 + 2][ln] + red[wn * 4 + 3][ln];
  const float inv = 8.0f / fmaxf(sqrtf(tot), 1e-12f);   // * W^0.5 (=8)
  float* yb = y + (size_t)b * CCH * NTOK;
#pragma unroll
  for (int mt = 0; mt < 4; ++mt)
#pragma unroll
    for (int r = 0; r < 8; ++r) {
      const int c = wm * 64 + mt * 16 + lh * 8 + r;
      yb[(size_t)c * NTOK + n_glob] = vals[mt][r] * inv * g_l[c];
    }
}

// ---------- launch ----------
extern "C" void kernel_launch(void* const* d_in, const int* in_sizes, int n_in,
                              void* d_out, int out_size, void* d_ws, size_t ws_size,
                              hipStream_t stream) {
  const float* x     = (const float*)d_in[0];
  const float* w_qkv = (const float*)d_in[1];
  const float* w_out = (const float*)d_in[2];
  const float* b_out = (const float*)d_in[3];
  const float* g     = (const float*)d_in[4];
  float* y = (float*)d_out;

  char* ws = (char*)d_ws;
  size_t off = 0;
  auto alloc = [&](size_t bytes) -> void* {
    void* p = ws + off;
    off = (off + bytes + 255) & ~(size_t)255;
    return p;
  };
  __bf16* wq_bf  = (__bf16*)alloc((size_t)3 * HIDDEN * CCH * 2);       // 768 KB
  __bf16* wo_bf  = (__bf16*)alloc((size_t)CCH * HIDDEN * 2);           // 256 KB
  __bf16* xT_bf  = (__bf16*)alloc((size_t)BATCH * NTOK * CCH * 2);     // 32 MB
  float*  qT_f   = (float*) alloc((size_t)BATCH * HIDDEN * NTOK * 4);  // 128 MB
  float*  k_f    = (float*) alloc((size_t)BATCH * HIDDEN * NTOK * 4);  // 128 MB
  __bf16* v_bf   = (__bf16*)alloc((size_t)BATCH * HIDDEN * NTOK * 2);  // 64 MB
  __bf16* k_bf   = (__bf16*)alloc((size_t)BATCH * HIDDEN * NTOK * 2);  // 64 MB
  __bf16* q_bf   = (__bf16*)alloc((size_t)BATCH * HIDDEN * NTOK * 2);  // 64 MB
  __bf16* ctx_bf = (__bf16*)alloc((size_t)BATCH * HEADS * DIMH * DIMH * 2);
  __bf16* out_bf = (__bf16*)alloc((size_t)BATCH * NTOK * HIDDEN * 2);  // 64 MB

  // weight conversions
  cvt_bf16_kernel<<<1536, 256, 0, stream>>>(w_qkv, wq_bf, 3 * HIDDEN * CCH);
  cvt_bf16_kernel<<<512, 256, 0, stream>>>(w_out, wo_bf, CCH * HIDDEN);

  // x -> xT bf16
  transpose_x_kernel<<<dim3(NTOK / 32, CCH / 32, BATCH), dim3(32, 8), 0, stream>>>(x, xT_bf);

  // GEMM1: qkv
  gemm_qkv_kernel<<<dim3(NTOK / 64, (3 * HIDDEN) / 128, BATCH), 256, 0, stream>>>(
      wq_bf, xT_bf, qT_f, k_f, v_bf);

  // softmaxes
  softmax_k_kernel<<<BATCH * HEADS * DIMH, 256, 0, stream>>>(k_f, k_bf);
  softmax_q_kernel<<<(BATCH * HEADS * NTOK) / 8, 256, 0, stream>>>(qT_f, q_bf);

  // GEMM2: context per (b,h)  (TDM double-buffered LDS staging)
  gemm_ctx_kernel<<<BATCH * HEADS, 128, 0, stream>>>(k_bf, v_bf, ctx_bf);

  // GEMM3: out per (b,h)
  gemm_out_kernel<<<dim3(NTOK / 128, BATCH * HEADS), 256, 0, stream>>>(ctx_bf, q_bf, out_bf);

  // GEMM4 + bias + channel L2-norm + g*sqrt(W)
  gemm_proj_norm_kernel<<<dim3(NTOK / 32, BATCH), 256, 0, stream>>>(
      wo_bf, out_bf, b_out, g, y);
}
